// AlignmentModule_13769665151409
// MI455X (gfx1250) — compile-verified, matrix operand order checked by reading, not run on hardware
//
#include <hip/hip_runtime.h>
#include <hip/hip_bf16.h>
#include <stdint.h>

// ---------------------------------------------------------------------------
// Problem constants (from reference): B=128, LC=256, LE=1024, D=256, M=512
// ---------------------------------------------------------------------------
#define BB  128
#define LCC 256
#define LEE 1024
#define DD  256
#define MM  512
#define NEGBIG (-1e9f)

typedef __bf16 bf16;
typedef __attribute__((ext_vector_type(16))) __bf16       v16bf;
typedef __attribute__((ext_vector_type(8)))  float        v8f;
typedef __attribute__((ext_vector_type(4)))  float        f32x4;
typedef __attribute__((ext_vector_type(4)))  unsigned int u32x4;
typedef __attribute__((ext_vector_type(4)))  int          i32x4;
typedef __attribute__((ext_vector_type(8)))  int          i32x8;

#ifndef __has_builtin
#define __has_builtin(x) 0
#endif
#if __has_builtin(__builtin_amdgcn_tensor_load_to_lds) && \
    __has_builtin(__builtin_amdgcn_s_wait_tensorcnt)
#define HAVE_TDM 1
#else
#define HAVE_TDM 0
#endif

union FragU { v16bf v; u32x4 u4[2]; };

// Load a 16x32 bf16 operand fragment from an LDS tile stored row-major with
// leading dimension `ld` (elements). Matches CDNA5 16-bit A/B operand layout:
// lanes 0-15 hold row (lane&15), K 0..7 and 16..23; lanes 16-31 hold the same
// row with K 8..15 and 24..31. Two 16-byte ds_load_b128 per fragment.
__device__ __forceinline__ v16bf load_frag(const bf16* p, int ld) {
  const int lane = threadIdx.x & 31;
  const int r = lane & 15;
  const int h = (lane >> 4) & 1;
  const bf16* q = p + r * ld + h * 8;
  FragU f;
  f.u4[0] = *(const u32x4*)(q);
  f.u4[1] = *(const u32x4*)(q + 16);
  return f.v;
}

__device__ __forceinline__ v8f wmma_bf16(v16bf a, v16bf b, v8f c) {
  // D = A(16x32 bf16) * B(32x16 bf16) + C(16x16 f32)
  return __builtin_amdgcn_wmma_f32_16x16x32_bf16(
      /*neg_a=*/false, a, /*neg_b=*/false, b,
      /*c_mod=*/(short)0, c, /*reuse_a=*/false, /*reuse_b=*/false);
}

#if HAVE_TDM
// Issue one TDM 2D tile load: global (row-major, bf16, stride0 elements per
// row) -> LDS, tile (tile_d0 x tile_d1) elements.  D# built per CDNA5 ISA
// ch.8: group0 = {count, lds_addr, global_addr, type=2}; group1 = {data_size,
// tensor dims, tile dims, dim0 stride}.  Tracked by TENSORcnt.
__device__ __forceinline__ void tdm_load_2d(unsigned lds_off, const bf16* gptr,
                                            int tile_d0, int tile_d1,
                                            int tensor_d0, int tensor_d1,
                                            int stride0) {
  unsigned long long ga = (unsigned long long)(uintptr_t)gptr;
  u32x4 g0;
  g0[0] = 1u;                                         // count=1 (valid user D#)
  g0[1] = lds_off;                                    // lds_addr (bytes)
  g0[2] = (unsigned)(ga & 0xffffffffu);               // global_addr[31:0]
  g0[3] = (unsigned)((ga >> 32) & 0x01ffffffu)        // global_addr[56:32]
        | (2u << 30);                                 // type = 2 ("image")
  i32x8 g1;
  g1[0] = (1 << 16);                                  // data_size = 1 -> 2B
  g1[1] = (int)(((unsigned)tensor_d0 & 0xffffu) << 16);        // td0[15:0]
  g1[2] = (int)(((unsigned)tensor_d0 >> 16) & 0xffffu)
        | (int)(((unsigned)tensor_d1 & 0xffffu) << 16);        // td0 hi | td1 lo
  g1[3] = (int)(((unsigned)tensor_d1 >> 16) & 0xffffu)
        | (int)(((unsigned)tile_d0 & 0xffffu) << 16);          // td1 hi | tile0
  g1[4] = tile_d1 & 0xffff;                                    // tile1 (tile2=0)
  g1[5] = stride0;                                             // dim0 stride lo
  g1[6] = 0;
  g1[7] = 0;
  i32x4 z4 = {0, 0, 0, 0};
#if defined(__clang_major__) && __clang_major__ >= 23
  i32x8 z8 = {0, 0, 0, 0, 0, 0, 0, 0};
  __builtin_amdgcn_tensor_load_to_lds(g0, g1, z4, z4, z8, 0);
#else
  __builtin_amdgcn_tensor_load_to_lds(g0, g1, z4, z4, 0);
#endif
}
#endif  // HAVE_TDM

// ---------------------------------------------------------------------------
// Elementwise fp32 -> bf16 cast, 8 elements (16B out) per thread
// ---------------------------------------------------------------------------
__global__ void cast_f32_bf16_kernel(const float* __restrict__ in,
                                     bf16* __restrict__ out, int n8) {
  int i = blockIdx.x * blockDim.x + threadIdx.x;
  if (i >= n8) return;
  f32x4 a = *(const f32x4*)&in[(size_t)i * 8];
  f32x4 b = *(const f32x4*)&in[(size_t)i * 8 + 4];
  union { u32x4 u; bf16 h[8]; } o;
#pragma unroll
  for (int t = 0; t < 4; ++t) { o.h[t] = (bf16)a[t]; o.h[4 + t] = (bf16)b[t]; }
  *(u32x4*)&out[(size_t)i * 8] = o.u;
}

__global__ void zero_f32_kernel(float* __restrict__ p, int n) {
  int i = blockIdx.x * blockDim.x + threadIdx.x;
  if (i < n) p[i] = 0.0f;
}

// ---------------------------------------------------------------------------
// Out[row0..+32, col0..+256) = relu(A(bf16, lda=K) @ W(f32, K x Ntot) + bias)
// cast to bf16.  8 waves: wave = (mt 0..1, cg 0..3); each wave owns a 16x64
// output slab (4 accumulator tiles).
// ---------------------------------------------------------------------------
__global__ __launch_bounds__(256) void gemm_bias_relu_kernel(
    const bf16* __restrict__ A, const float* __restrict__ W,
    const float* __restrict__ bias, bf16* __restrict__ Out,
    int K, int Ntot) {
  __shared__ __align__(16) bf16 sA[32 * 32];
  __shared__ __align__(16) bf16 sWT[256 * 32];  // [n][k] slice (W transposed)
  const int row0 = blockIdx.x * 32;
  const int col0 = blockIdx.y * 256;
  const int tid = threadIdx.x;
  const int wave = tid >> 5, lane = tid & 31;
  const int lr = lane & 15, lh = lane >> 4;
  const int mt = wave >> 2, cg = wave & 3;
  v8f acc[4] = {};
  for (int k0 = 0; k0 < K; k0 += 32) {
    // A tile: 32x32 bf16 = 128 x (8-elem b128 groups)
    if (tid < 128) {
      int r = tid >> 2, c8 = (tid & 3) * 8;
      *(u32x4*)&sA[r * 32 + c8] =
          *(const u32x4*)&A[(size_t)(row0 + r) * K + k0 + c8];
    }
    // W^T slice: one f32x4 global load (4 consecutive n), 4 scalar LDS stores
    for (int g = tid; g < 32 * 64; g += 256) {
      int c = g >> 6;                // k within slice
      int n0 = (g & 63) * 4;         // n start
      f32x4 w = *(const f32x4*)&W[(size_t)(k0 + c) * Ntot + col0 + n0];
#pragma unroll
      for (int t = 0; t < 4; ++t) sWT[(n0 + t) * 32 + c] = (bf16)w[t];
    }
    __syncthreads();
    v16bf a = load_frag(sA + mt * 16 * 32, 32);
#pragma unroll
    for (int t = 0; t < 4; ++t) {
      v16bf bb = load_frag(sWT + (cg * 64 + t * 16) * 32, 32);
      acc[t] = wmma_bf16(a, bb, acc[t]);
    }
    __syncthreads();
  }
#pragma unroll
  for (int t = 0; t < 4; ++t) {
    int col = col0 + cg * 64 + t * 16 + lr;
    float bv = bias[col];
#pragma unroll
    for (int r = 0; r < 8; ++r) {
      int row = row0 + mt * 16 + lh * 8 + r;
      float v = acc[t][r] + bv;
      Out[(size_t)row * Ntot + col] = (bf16)(v > 0.0f ? v : 0.0f);
    }
  }
}

// ---------------------------------------------------------------------------
// Flash attention pass (D = 256 fixed):
//   O[b,q,:] = softmax_kv(Q[b]K[b]^T + (1-mask_kv)*NEG) @ V[b]
// Block: 32 Q rows, KV streamed in tiles of 64 with online softmax.
// Q block and K tiles arrive via the Tensor Data Mover when available.
// LDS: sQ 16KB + sKV 32KB (K tile, then V^T tile) + sS 8KB + sP 4KB  (<64KB)
// ---------------------------------------------------------------------------
__global__ __launch_bounds__(256) void flash_attn_kernel(
    const bf16* __restrict__ Q, const bf16* __restrict__ Kg,
    const bf16* __restrict__ V, const int* __restrict__ mkv,
    bf16* __restrict__ O, int Lq, int Lkv) {
  __shared__ __align__(16) bf16 sQ[32 * 256];
  __shared__ __align__(16) bf16 sKV[64 * 256];   // K tile, then V^T [d][kv]
  __shared__ __align__(16) float sS[32 * 64];
  __shared__ __align__(16) bf16 sP[32 * 64];
  __shared__ float sMax[32];
  __shared__ float sSum[32];
  __shared__ float sScale[32];

  const int b = blockIdx.y;
  const int q0 = blockIdx.x * 32;
  const int tid = threadIdx.x;
  const int wave = tid >> 5, lane = tid & 31;
  const int lr = lane & 15, lh = lane >> 4;
  const int mt = wave >> 2;   // 0..1: Q-row tile
  const int cg = wave & 3;    // 0..3: col group

  const bf16* Qb = Q + (size_t)b * Lq * 256;
  const bf16* Kb = Kg + (size_t)b * Lkv * 256;
  const bf16* Vb = V + (size_t)b * Lkv * 256;
  const int* mb = mkv + (size_t)b * Lkv;

  // ---- Q block load (32 x 256, contiguous) ----
#if HAVE_TDM
  if (wave == 0) {
    tdm_load_2d((unsigned)(uintptr_t)&sQ[0], Qb + (size_t)q0 * 256,
                256, 32, 256, 32, 256);
    __builtin_amdgcn_s_wait_tensorcnt(0);
  }
#else
  for (int g = tid; g < 32 * 32; g += 256)
    *(u32x4*)&sQ[g * 8] = *(const u32x4*)&Qb[(size_t)q0 * 256 + g * 8];
#endif
  if (tid < 32) { sMax[tid] = -3.0e38f; sSum[tid] = 0.0f; }
  v8f accO[4] = {};
  __syncthreads();

  for (int j0 = 0; j0 < Lkv; j0 += 64) {
    // ---- phase 1: K tile (64 x 256, row-major, contiguous) ----
#if HAVE_TDM
    if (wave == 0) {
      tdm_load_2d((unsigned)(uintptr_t)&sKV[0], Kb + (size_t)j0 * 256,
                  256, 64, 256, 64, 256);
      __builtin_amdgcn_s_wait_tensorcnt(0);
    }
#else
    for (int g = tid; g < 64 * 32; g += 256)
      *(u32x4*)&sKV[g * 8] = *(const u32x4*)&Kb[(size_t)j0 * 256 + g * 8];
#endif
    __syncthreads();

    // ---- phase 2: S = Q @ K^T (each wave one 16x16 tile), mask, stash ----
    v8f s = {};
#pragma unroll
    for (int k0 = 0; k0 < 256; k0 += 32) {
      v16bf a  = load_frag(sQ + mt * 16 * 256 + k0, 256);
      v16bf bb = load_frag(sKV + cg * 16 * 256 + k0, 256);
      s = wmma_bf16(a, bb, s);
    }
    {
      float mbias = (1.0f - (float)mb[j0 + cg * 16 + lr]) * NEGBIG;
#pragma unroll
      for (int r = 0; r < 8; ++r) s[r] += mbias;
    }
#pragma unroll
    for (int r = 0; r < 8; ++r)
      sS[(mt * 16 + lh * 8 + r) * 64 + cg * 16 + lr] = s[r];
    __syncthreads();

    // ---- phase 3: online-softmax stats (threads 0..31) + V^T load ----
    // V^T: one b128 global load (8 d-elems of one kv row), 8 scattered stores
    for (int g = tid; g < 64 * 32; g += 256) {
      int r = g >> 5;               // kv row
      int c8 = (g & 31) * 8;        // d start
      u32x4 v = *(const u32x4*)&Vb[(size_t)(j0 + r) * 256 + c8];
      const bf16* pv = (const bf16*)&v;
#pragma unroll
      for (int t = 0; t < 8; ++t) sKV[(c8 + t) * 64 + r] = pv[t];
    }
    if (tid < 32) {
      float mOld = sMax[tid];
      float mNew = mOld;
      const float* sr = sS + tid * 64;
      for (int c = 0; c < 64; ++c) mNew = fmaxf(mNew, sr[c]);
      float scale = __expf(mOld - mNew);
      float sum = 0.0f;
      bf16* pr = sP + tid * 64;
      for (int c = 0; c < 64; ++c) {
        float p = __expf(sr[c] - mNew);
        pr[c] = (bf16)p;
        sum += p;
      }
      sSum[tid] = sSum[tid] * scale + sum;
      sMax[tid] = mNew;
      sScale[tid] = scale;
    }
    __syncthreads();

    // ---- phase 4: rescale O, accumulate P @ V (wave: 16 rows x 64 d-cols) --
#pragma unroll
    for (int t = 0; t < 4; ++t)
#pragma unroll
      for (int r = 0; r < 8; ++r)
        accO[t][r] *= sScale[mt * 16 + lh * 8 + r];
    v16bf pa0 = load_frag(sP + mt * 16 * 64, 64);       // P, K(kv) 0..31
    v16bf pa1 = load_frag(sP + mt * 16 * 64 + 32, 64);  // P, K(kv) 32..63
#pragma unroll
    for (int t = 0; t < 4; ++t) {
      int d0 = (cg * 64 + t * 16) * 64;                 // V^T row block
      v16bf b0 = load_frag(sKV + d0, 64);
      v16bf b1 = load_frag(sKV + d0 + 32, 64);
      accO[t] = wmma_bf16(pa0, b0, accO[t]);
      accO[t] = wmma_bf16(pa1, b1, accO[t]);
    }
    __syncthreads();
  }

  // ---- epilogue: normalize and store ----
  bf16* Ob = O + (size_t)b * Lq * 256;
#pragma unroll
  for (int t = 0; t < 4; ++t) {
    int col = cg * 64 + t * 16 + lr;
#pragma unroll
    for (int r = 0; r < 8; ++r) {
      int row = q0 + mt * 16 + lh * 8 + r;
      float inv = 1.0f / sSum[mt * 16 + lh * 8 + r];
      Ob[(size_t)row * 256 + col] = (bf16)(accO[t][r] * inv);
    }
  }
}

// ---------------------------------------------------------------------------
// R[b,:] += sum_rows relu( [Att | X] @ Wr + br ) over a 32-row slab.
// K = 512: first 256 from Att (Wr rows 0..255), next 256 from X (rows 256..511)
// ---------------------------------------------------------------------------
__global__ __launch_bounds__(256) void rsum_kernel(
    const bf16* __restrict__ Att, const bf16* __restrict__ X,
    const float* __restrict__ Wr, const float* __restrict__ br,
    float* __restrict__ Rout, int L) {
  __shared__ __align__(16) bf16 sA[32 * 32];
  __shared__ __align__(16) bf16 sWT[256 * 32];
  const int b = blockIdx.y;
  const int r0 = blockIdx.x * 32;
  const int tid = threadIdx.x;
  const int wave = tid >> 5, lane = tid & 31;
  const int lr = lane & 15, lh = lane >> 4;
  const int mt = wave >> 2, cg = wave & 3;
  const bf16* Ab = Att + (size_t)b * L * 256;
  const bf16* Xb = X + (size_t)b * L * 256;
  v8f acc[4] = {};
  for (int k0 = 0; k0 < 512; k0 += 32) {
    const bf16* src = (k0 < 256) ? Ab : Xb;
    const int kk = k0 & 255;
    if (tid < 128) {
      int r = tid >> 2, c8 = (tid & 3) * 8;
      *(u32x4*)&sA[r * 32 + c8] =
          *(const u32x4*)&src[(size_t)(r0 + r) * 256 + kk + c8];
    }
    for (int g = tid; g < 32 * 64; g += 256) {
      int c = g >> 6;
      int n0 = (g & 63) * 4;
      f32x4 w = *(const f32x4*)&Wr[(size_t)(k0 + c) * 256 + n0];
#pragma unroll
      for (int t = 0; t < 4; ++t) sWT[(n0 + t) * 32 + c] = (bf16)w[t];
    }
    __syncthreads();
    v16bf a = load_frag(sA + mt * 16 * 32, 32);
#pragma unroll
    for (int t = 0; t < 4; ++t) {
      v16bf bb = load_frag(sWT + (cg * 64 + t * 16) * 32, 32);
      acc[t] = wmma_bf16(a, bb, acc[t]);
    }
    __syncthreads();
  }
  // bias + relu + per-lane row sum, then one f32 atomic per (lane, tile)
#pragma unroll
  for (int t = 0; t < 4; ++t) {
    int col = cg * 64 + t * 16 + lr;
    float bv = br[col];
    float sum = 0.0f;
#pragma unroll
    for (int r = 0; r < 8; ++r) {
      float v = acc[t][r] + bv;
      sum += (v > 0.0f ? v : 0.0f);
    }
    atomicAdd(&Rout[(size_t)b * 256 + col], sum);
  }
}

// ---------------------------------------------------------------------------
// m = [r1, r2, r1*r2, r1-r2] -> bf16 (B x 1024)
// ---------------------------------------------------------------------------
__global__ void build_m_kernel(const float* __restrict__ r1,
                               const float* __restrict__ r2,
                               bf16* __restrict__ m) {
  int i = blockIdx.x * blockDim.x + threadIdx.x;
  if (i >= BB * 1024) return;
  int b = i >> 10, j = i & 1023;
  float a = r1[b * 256 + (j & 255)];
  float c = r2[b * 256 + (j & 255)];
  int seg = j >> 8;
  float v = (seg == 0) ? a : (seg == 1) ? c : (seg == 2) ? a * c : a - c;
  m[i] = (bf16)v;
}

// ---------------------------------------------------------------------------
// out = h(bf16, B x 512) @ Wo(512 x 3) + bo     (tiny; plain f32 VALU)
// ---------------------------------------------------------------------------
__global__ void final_out_kernel(const bf16* __restrict__ h,
                                 const float* __restrict__ Wo,
                                 const float* __restrict__ bo,
                                 float* __restrict__ out) {
  int i = blockIdx.x * blockDim.x + threadIdx.x;
  if (i >= BB * 3) return;
  int b = i / 3, j = i % 3;
  float s = bo[j];
  const bf16* hb = h + (size_t)b * MM;
  for (int k = 0; k < MM; ++k) s += (float)hb[k] * Wo[k * 3 + j];
  out[i] = s;
}

// ---------------------------------------------------------------------------
// Host launcher
// ---------------------------------------------------------------------------
extern "C" void kernel_launch(void* const* d_in, const int* in_sizes, int n_in,
                              void* d_out, int out_size, void* d_ws, size_t ws_size,
                              hipStream_t stream) {
  (void)in_sizes; (void)n_in; (void)out_size; (void)ws_size;
  const float* criteria = (const float*)d_in[0];   // (B, LC, D)
  const float* ehr      = (const float*)d_in[1];   // (B, LE, D)
  const int*   cmask    = (const int*)d_in[2];     // (B, LC)
  const int*   emask    = (const int*)d_in[3];     // (B, LE)
  const float* Wa       = (const float*)d_in[4];   // (D, D)
  const float* ba       = (const float*)d_in[5];   // (D,)
  const float* Wr       = (const float*)d_in[6];   // (2D, D)
  const float* br       = (const float*)d_in[7];   // (D,)
  const float* Wm       = (const float*)d_in[8];   // (4D, M)
  const float* bm       = (const float*)d_in[9];   // (M,)
  const float* Wo       = (const float*)d_in[10];  // (M, 3)
  const float* bo       = (const float*)d_in[11];  // (3,)
  float* out = (float*)d_out;                      // (B, 3)

  // Workspace carve-up (~252 MB total)
  char* ws = (char*)d_ws;
  size_t off = 0;
  auto carve = [&](size_t bytes) -> char* {
    char* p = ws + off;
    off += (bytes + 255) & ~(size_t)255;
    return p;
  };
  bf16* cr_bf = (bf16*)carve((size_t)BB * LCC * DD * 2);  // criteria bf16
  bf16* eh_bf = (bf16*)carve((size_t)BB * LEE * DD * 2);  // ehr bf16
  bf16* c_bf  = (bf16*)carve((size_t)BB * LCC * DD * 2);  // relu(crit@Wa+ba)
  bf16* e_bf  = (bf16*)carve((size_t)BB * LEE * DD * 2);  // relu(ehr@Wa+ba)
  bf16* attc  = (bf16*)carve((size_t)BB * LCC * DD * 2);
  bf16* atte  = (bf16*)carve((size_t)BB * LEE * DD * 2);
  float* r1   = (float*)carve((size_t)BB * DD * 4);
  float* r2   = (float*)carve((size_t)BB * DD * 4);
  bf16* m_bf  = (bf16*)carve((size_t)BB * 4 * DD * 2);
  bf16* h_bf  = (bf16*)carve((size_t)BB * MM * 2);

  const int nCrit = BB * LCC * DD;       // 8,388,608
  const int nEhr  = BB * LEE * DD;       // 33,554,432

  // 1) bf16 casts of the raw inputs (V matrices + concat halves)
  cast_f32_bf16_kernel<<<(nCrit / 8 + 255) / 256, 256, 0, stream>>>(
      criteria, cr_bf, nCrit / 8);
  cast_f32_bf16_kernel<<<(nEhr / 8 + 255) / 256, 256, 0, stream>>>(
      ehr, eh_bf, nEhr / 8);

  // 2) projections: c = relu(criteria@Wa+ba), e = relu(ehr@Wa+ba)
  gemm_bias_relu_kernel<<<dim3(BB * LCC / 32, 1), 256, 0, stream>>>(
      cr_bf, Wa, ba, c_bf, DD, DD);
  gemm_bias_relu_kernel<<<dim3(BB * LEE / 32, 1), 256, 0, stream>>>(
      eh_bf, Wa, ba, e_bf, DD, DD);

  // 3) att_c = softmax_Le(c e^T, mask=ehr_mask) @ ehr   (flash, Q=c K=e V=ehr)
  flash_attn_kernel<<<dim3(LCC / 32, BB), 256, 0, stream>>>(
      c_bf, e_bf, eh_bf, emask, attc, LCC, LEE);
  // 4) att_e = softmax_Lc(e c^T, mask=crit_mask) @ criteria (Q=e K=c V=crit)
  flash_attn_kernel<<<dim3(LEE / 32, BB), 256, 0, stream>>>(
      e_bf, c_bf, cr_bf, cmask, atte, LEE, LCC);

  // 5) r1 / r2: row-summed relu([att|x] @ Wr + br)
  zero_f32_kernel<<<(BB * DD + 255) / 256, 256, 0, stream>>>(r1, BB * DD);
  zero_f32_kernel<<<(BB * DD + 255) / 256, 256, 0, stream>>>(r2, BB * DD);
  rsum_kernel<<<dim3(LCC / 32, BB), 256, 0, stream>>>(attc, cr_bf, Wr, br, r1, LCC);
  rsum_kernel<<<dim3(LEE / 32, BB), 256, 0, stream>>>(atte, eh_bf, Wr, br, r2, LEE);

  // 6) m = [r1, r2, r1*r2, r1-r2];  h = relu(m @ Wm + bm);  out = h @ Wo + bo
  build_m_kernel<<<(BB * 1024 + 255) / 256, 256, 0, stream>>>(r1, r2, m_bf);
  gemm_bias_relu_kernel<<<dim3(BB / 32, MM / 256), 256, 0, stream>>>(
      m_bf, Wm, bm, h_bf, 4 * DD, MM);
  final_out_kernel<<<(BB * 3 + 255) / 256, 256, 0, stream>>>(h_bf, Wo, bo, out);
}
